// TurboQuantWrapper_16733192585227
// MI455X (gfx1250) — matrix operand order repaired and checked by reading, not compiled, not run on hardware
//
#include <hip/hip_runtime.h>
#include <hip/hip_bf16.h>
#include <stdint.h>

// ---------------------------------------------------------------------------
// O=11008, I=4096, B=2048, G=128, N=32.
// out[b,o] = sum_k rotate(x)[b,k] * Wq[o,k],
//   Wq[o,k] = centroids[idx(o,k)] * norms[o, k/128] / sqrt(128)
// 184.7 GFLOP over ~142 MB HBM traffic -> compute bound -> f16 WMMA, f32 acc.
// Double-buffered LDS pipeline; buffer selection via integer offsets into a
// single __shared__ array so fragment loads stay ds_load_b128 (not flat).
// ---------------------------------------------------------------------------

#define O_DIM 11008
#define I_DIM 4096
#define B_DIM 2048
#define G_DIM 128
#define N_GRP 32

#define BM 128           // M tile (batch rows)
#define BN 128           // N tile (output rows)
#define BK 64            // K tile per LDS stage
#define SSTR 72          // padded LDS row stride in f16 (144B)
#define NSTAGE (I_DIM / BK)
#define ABUF  (BM * SSTR)       // elements per A buffer
#define BBUF  (BN * SSTR)       // elements per B buffer

typedef __attribute__((ext_vector_type(16))) _Float16 v16h;
typedef __attribute__((ext_vector_type(8)))  float    v8f;

typedef __attribute__((address_space(3))) unsigned char lds_byte_t;

static __device__ __forceinline__ unsigned lds_off(void* p) {
    // generic -> LDS addrspace, ptrtoint gives the wave-relative LDS byte
    // address required by the async-to-LDS instructions (VDST operand).
    return (unsigned)(unsigned long long)(lds_byte_t*)p;
}

static __device__ __forceinline__ unsigned short f2h_bits(float f) {
    _Float16 h = (_Float16)f;
    return __builtin_bit_cast(unsigned short, h);
}

static __device__ __forceinline__ v16h ld_frag(const _Float16* p0, const _Float16* p1) {
    union { v16h h; uint4 q[2]; } u;
    u.q[0] = *reinterpret_cast<const uint4*>(p0);
    u.q[1] = *reinterpret_cast<const uint4*>(p1);
    return u.h;
}

static __device__ __forceinline__ void wait_asynccnt0() {
#if __has_builtin(__builtin_amdgcn_s_wait_asynccnt)
    __builtin_amdgcn_s_wait_asynccnt(0);
#else
    asm volatile("s_wait_asynccnt 0" ::: "memory");
#endif
}

// 64B/thread async memory->LDS copy; INST_OFFSET applies to both addresses.
static __device__ __forceinline__ void async_copy64(unsigned lds, unsigned long long ga) {
    asm volatile(
        "global_load_async_to_lds_b128 %0, %1, off\n\t"
        "global_load_async_to_lds_b128 %0, %1, off offset:16\n\t"
        "global_load_async_to_lds_b128 %0, %1, off offset:32\n\t"
        "global_load_async_to_lds_b128 %0, %1, off offset:48"
        :: "v"(lds), "v"(ga) : "memory");
}

// Unpack 32 3-bit codes (12B in ww0..2) through a pre-scaled f16 table packed
// in two 64-bit regs; emit 4 x uint4 (ds_store_b128) to LDS.
static __device__ __forceinline__ void dequant32(
        unsigned ww0, unsigned ww1, unsigned ww2,
        unsigned long long tabLo, unsigned long long tabHi, uint4* dst) {
    unsigned int ww[3] = { ww0, ww1, ww2 };
#pragma unroll
    for (int gi = 0; gi < 4; ++gi) {
        const int bb = gi * 3;
        const unsigned int b0 = (ww[(bb    ) >> 2] >> (((bb    ) & 3) * 8)) & 0xFF;
        const unsigned int b1 = (ww[(bb + 1) >> 2] >> (((bb + 1) & 3) * 8)) & 0xFF;
        const unsigned int b2 = (ww[(bb + 2) >> 2] >> (((bb + 2) & 3) * 8)) & 0xFF;
        unsigned int idx[8];
        idx[0] =  b0        & 7;
        idx[1] = (b0 >> 3)  & 7;
        idx[2] = ((b0 >> 6) | (b1 << 2)) & 7;
        idx[3] = (b1 >> 1)  & 7;
        idx[4] = (b1 >> 4)  & 7;
        idx[5] = ((b1 >> 7) | (b2 << 1)) & 7;
        idx[6] = (b2 >> 2)  & 7;
        idx[7] = (b2 >> 5)  & 7;
        unsigned int dw[4];
#pragma unroll
        for (int e = 0; e < 8; e += 2) {
            unsigned long long t0 = (idx[e]     & 4) ? tabHi : tabLo;
            unsigned long long t1 = (idx[e + 1] & 4) ? tabHi : tabLo;
            unsigned int lo = (unsigned int)((t0 >> ((idx[e]     & 3) * 16)) & 0xFFFF);
            unsigned int hi = (unsigned int)((t1 >> ((idx[e + 1] & 3) * 16)) & 0xFFFF);
            dw[e >> 1] = lo | (hi << 16);
        }
        dst[gi] = make_uint4(dw[0], dw[1], dw[2], dw[3]);
    }
}

// ---------------------------------------------------------------------------
// Kernel 1: rx = s2 * FWHT(x * s1) / sqrt(128), emitted as f16.
// ---------------------------------------------------------------------------
__global__ __launch_bounds__(256) void rotate_fwht_kernel(
        const float* __restrict__ x,
        const float* __restrict__ s1,
        const float* __restrict__ s2,
        _Float16* __restrict__ rxh) {
    const int lane = threadIdx.x & 31;
    const int wave = threadIdx.x >> 5;
    const int g    = blockIdx.x * 8 + wave;
    const size_t base = (size_t)g * G_DIM;

    float v[4];
#pragma unroll
    for (int j = 0; j < 4; ++j)
        v[j] = x[base + lane + 32 * j] * s1[lane + 32 * j];

#pragma unroll
    for (int m = 1; m < 32; m <<= 1) {
#pragma unroll
        for (int j = 0; j < 4; ++j) {
            float t = __shfl_xor(v[j], m, 32);
            v[j] = (lane & m) ? (t - v[j]) : (v[j] + t);
        }
    }
    {
        float t0 = v[0] + v[1], t1 = v[0] - v[1];
        float t2 = v[2] + v[3], t3 = v[2] - v[3];
        v[0] = t0 + t2; v[2] = t0 - t2;
        v[1] = t1 + t3; v[3] = t1 - t3;
    }
    const float inv_sqrt_g = 0.08838834764831845f;
#pragma unroll
    for (int j = 0; j < 4; ++j)
        rxh[base + lane + 32 * j] = (_Float16)(v[j] * s2[lane + 32 * j] * inv_sqrt_g);
}

// ---------------------------------------------------------------------------
// Kernel 2: double-buffered WMMA GEMM.
// ---------------------------------------------------------------------------
__global__ __launch_bounds__(256) void quant_gemm_kernel(
        const _Float16*      __restrict__ rxh,     // (B, I) f16 rotated acts
        const unsigned char* __restrict__ pw,      // (O*N, 48) packed 3-bit
        const float*         __restrict__ norms,   // (O, N)
        const float*         __restrict__ cent,    // (8,)
        float*               __restrict__ out) {   // (B, O)
    __shared__ _Float16 sA[2 * ABUF];
    __shared__ _Float16 sB[2 * BBUF];

    const int tid  = threadIdx.x;
    const int lane = tid & 31;
    const int wave = tid >> 5;
    const int wm   = wave & 3;
    const int wn   = wave >> 2;
    const int sel  = lane >> 4;
    const int l15  = lane & 15;

    const int mBase = blockIdx.y * BM;
    const int nBase = blockIdx.x * BN;

    float cf[8];
#pragma unroll
    for (int i = 0; i < 8; ++i) cf[i] = cent[i];
    const float inv_sqrt_g = 0.08838834764831845f;

    // staging roles: 2 threads per row, 32 K (64B) each
    const int rowL = tid >> 1;
    const int half = tid & 1;
    const _Float16*      gA = rxh + (size_t)(mBase + rowL) * I_DIM + half * 32;
    const unsigned char* gW = pw  + (size_t)(nBase + rowL) * 1536 + half * 12;
    const float*         gN = norms + (size_t)(nBase + rowL) * N_GRP;

    const int      wOff    = rowL * SSTR + half * 32;     // writer offset (elems)
    const unsigned wA_lds0 = lds_off(&sA[wOff]);          // buffer0 byte addr
    const int      aBase   = (wm * 32 + l15) * SSTR + sel * 8;
    const int      bBase   = (wn * 64 + l15) * SSTR + sel * 8;

    v8f acc[2][4];
#pragma unroll
    for (int mt = 0; mt < 2; ++mt)
#pragma unroll
        for (int nt = 0; nt < 4; ++nt)
#pragma unroll
            for (int r = 0; r < 8; ++r) acc[mt][nt][r] = 0.0f;

    // ---- prologue: fill buffer 0 with stage 0 -----------------------------
    {
        async_copy64(wA_lds0, (unsigned long long)(size_t)(gA + 0));
        unsigned int w0 = *reinterpret_cast<const unsigned int*>(gW);
        unsigned int w1 = *reinterpret_cast<const unsigned int*>(gW + 4);
        unsigned int w2 = *reinterpret_cast<const unsigned int*>(gW + 8);
        const float scale = gN[0] * inv_sqrt_g;
        unsigned long long tabLo = 0ull, tabHi = 0ull;
#pragma unroll
        for (int i = 0; i < 4; ++i) {
            tabLo |= (unsigned long long)f2h_bits(cf[i]     * scale) << (16 * i);
            tabHi |= (unsigned long long)f2h_bits(cf[i + 4] * scale) << (16 * i);
        }
        dequant32(w0, w1, w2, tabLo, tabHi, reinterpret_cast<uint4*>(&sB[wOff]));
    }
    wait_asynccnt0();
    __syncthreads();

    // ---- main loop: one barrier per stage ---------------------------------
    for (int ks = 0; ks < NSTAGE; ++ks) {
        const int buf = ks & 1;
        const int nxt = buf ^ 1;
        const bool haveNext = (ks + 1 < NSTAGE);

        // issue stage ks+1 fetches before compute (latency hidden by WMMAs)
        unsigned int nw0 = 0, nw1 = 0, nw2 = 0;
        float nscale = 0.0f;
        if (haveNext) {
            const int k1 = (ks + 1) * BK;
            async_copy64(wA_lds0 + (unsigned)nxt * (ABUF * 2),
                         (unsigned long long)(size_t)(gA + k1));
            const unsigned char* wp = gW + (k1 >> 3) * 3;
            nw0 = *reinterpret_cast<const unsigned int*>(wp);
            nw1 = *reinterpret_cast<const unsigned int*>(wp + 4);
            nw2 = *reinterpret_cast<const unsigned int*>(wp + 8);
            nscale = gN[k1 >> 7] * inv_sqrt_g;
            if (ks + 2 < NSTAGE) {          // gfx1250 prefetch hints
                const int k2 = (ks + 2) * BK;
                __builtin_prefetch(gW + (k2 >> 3) * 3, 0, 3);
                __builtin_prefetch(gA + k2, 0, 3);
            }
        }

        // compute stage ks: integer buffer offsets keep these ds_load_b128
        const int aOff = aBase + buf * ABUF;
        const int bOff = bBase + buf * BBUF;
#pragma unroll
        for (int kk = 0; kk < BK; kk += 32) {
            v16h af[2], bf[4];
#pragma unroll
            for (int mt = 0; mt < 2; ++mt) {
                const _Float16* p = &sA[aOff + mt * 16 * SSTR + kk];
                af[mt] = ld_frag(p, p + 16);
            }
#pragma unroll
            for (int nt = 0; nt < 4; ++nt) {
                const _Float16* p = &sB[bOff + nt * 16 * SSTR + kk];
                bf[nt] = ld_frag(p, p + 16);
            }
#pragma unroll
            for (int mt = 0; mt < 2; ++mt)
#pragma unroll
                for (int nt = 0; nt < 4; ++nt)
                    acc[mt][nt] = __builtin_amdgcn_wmma_f32_16x16x32_f16(
                        false, af[mt], false, bf[nt],
                        (short)0, acc[mt][nt], false, false);
        }

        // dequant stage ks+1 into the other buffer (ww loads now resolved)
        if (haveNext) {
            unsigned long long tabLo = 0ull, tabHi = 0ull;
#pragma unroll
            for (int i = 0; i < 4; ++i) {
                tabLo |= (unsigned long long)f2h_bits(cf[i]     * nscale) << (16 * i);
                tabHi |= (unsigned long long)f2h_bits(cf[i + 4] * nscale) << (16 * i);
            }
            dequant32(nw0, nw1, nw2, tabLo, tabHi,
                      reinterpret_cast<uint4*>(&sB[wOff + nxt * BBUF]));
        }

        wait_asynccnt0();       // stage ks+1 A tile landed
        __syncthreads();        // all reads of `buf` + writes of `nxt` done
    }

    // ---- epilogue: C/D layout -> out[b*O + o] ------------------------------
#pragma unroll
    for (int mt = 0; mt < 2; ++mt) {
#pragma unroll
        for (int nt = 0; nt < 4; ++nt) {
            const int col   = nBase + wn * 64 + nt * 16 + l15;
            const int rbase = mBase + wm * 32 + mt * 16 + sel * 8;
            float* op = out + (size_t)rbase * O_DIM + col;
#pragma unroll
            for (int r = 0; r < 8; ++r)
                op[(size_t)r * O_DIM] = acc[mt][nt][r];
        }
    }
}

// ---------------------------------------------------------------------------
extern "C" void kernel_launch(void* const* d_in, const int* in_sizes, int n_in,
                              void* d_out, int out_size, void* d_ws, size_t ws_size,
                              hipStream_t stream) {
    (void)in_sizes; (void)n_in; (void)out_size; (void)ws_size;
    const float*         x     = (const float*)d_in[0];
    const unsigned char* pw    = (const unsigned char*)d_in[1];
    const float*         norms = (const float*)d_in[2];
    const float*         cent  = (const float*)d_in[3];
    const float*         s1    = (const float*)d_in[4];
    const float*         s2    = (const float*)d_in[5];
    float*               outp  = (float*)d_out;
    _Float16*            rxh   = (_Float16*)d_ws;

    rotate_fwht_kernel<<<(B_DIM * N_GRP) / 8, 256, 0, stream>>>(x, s1, s2, rxh);
    quant_gemm_kernel<<<dim3(O_DIM / BN, B_DIM / BM), 256, 0, stream>>>(
        rxh, pw, norms, cent, outp);
}
